// TransformerBlock_2130303779033
// MI455X (gfx1250) — matrix operand order repaired and checked by reading, not compiled, not run on hardware
//
#include <hip/hip_runtime.h>

// ---------------------------------------------------------------------------
// MI455X (gfx1250) transformer block: LN -> QKV -> flash-attn -> O+res ->
// LN -> FFN(GELU) -> +res.  All GEMMs via v_wmma_f32_16x16x32_f16 (wave32).
// Tile staging uses GLOBAL_LOAD_ASYNC_TO_LDS_B128 + s_wait_asynccnt with
// double buffering (builtin confirmed present on this toolchain).
// ---------------------------------------------------------------------------

typedef __attribute__((ext_vector_type(16))) _Float16 v16h;
typedef __attribute__((ext_vector_type(8)))  float    v8f;
typedef __attribute__((ext_vector_type(4)))  unsigned int uivec4;
typedef int v4i_vs __attribute__((vector_size(16)));   // builtin's pointee type

union FragU { v16h h; uivec4 u[2]; };

#define D_MODEL 1024
#define HEADS   16
#define HD      64
#define SEQ     2048
#define BATCH   2
#define ROWS    (BATCH * SEQ)   // 4096
#define D_FF    4096
#define LN_EPS  1e-6f

// ---------------- CDNA5 async global->LDS copy (16B per lane) ---------------
#if defined(__has_builtin)
#if __has_builtin(__builtin_amdgcn_global_load_async_to_lds_b128)
#define HAVE_ASYNC_LDS 1
#endif
#if __has_builtin(__builtin_amdgcn_s_wait_asynccnt)
#define HAVE_WAIT_ASYNC 1
#endif
#endif

#ifndef HAVE_ASYNC_LDS
#define HAVE_ASYNC_LDS 0
#warning "gfx1250 global_load_async_to_lds builtin not found: using sync LDS staging"
#endif
#ifndef HAVE_WAIT_ASYNC
#define HAVE_WAIT_ASYNC 0
#endif

#define AS1 __attribute__((address_space(1)))
#define AS3 __attribute__((address_space(3)))

__device__ __forceinline__ void async_cp16(const _Float16* g, _Float16* l) {
#if HAVE_ASYNC_LDS
  __builtin_amdgcn_global_load_async_to_lds_b128(
      (AS1 v4i_vs*)g, (AS3 v4i_vs*)l, 0, 0);
#else
  *(uivec4*)l = *(const uivec4*)g;   // sync fallback: vmem -> vgpr -> ds
#endif
}

__device__ __forceinline__ void wait_async_all() {
#if HAVE_ASYNC_LDS
#if HAVE_WAIT_ASYNC
  __builtin_amdgcn_s_wait_asynccnt(0);
#else
  asm volatile("s_wait_asynccnt 0" ::: "memory");
#endif
#endif
}

// ---------------------------------------------------------------------------
// fp32 -> f16 conversion (weights)
// ---------------------------------------------------------------------------
__global__ void cvt_f32_f16(const float* __restrict__ s,
                            _Float16* __restrict__ d, int n) {
  for (int i = blockIdx.x * blockDim.x + threadIdx.x; i < n;
       i += gridDim.x * blockDim.x)
    d[i] = (_Float16)s[i];
}

// ---------------------------------------------------------------------------
// LayerNorm over last dim (D_MODEL), one block per row, f16 output
// ---------------------------------------------------------------------------
__global__ __launch_bounds__(256)
void layernorm_f16(const float* __restrict__ x, const float* __restrict__ gam,
                   const float* __restrict__ bet, _Float16* __restrict__ out) {
  const int row = blockIdx.x, tid = threadIdx.x;
  const float* xr = x + (size_t)row * D_MODEL;
  float s = 0.f, s2 = 0.f;
  #pragma unroll
  for (int i = tid; i < D_MODEL; i += 256) {
    float v = xr[i]; s += v; s2 += v * v;
  }
  __shared__ float r1[256], r2[256];
  r1[tid] = s; r2[tid] = s2;
  __syncthreads();
  for (int st = 128; st > 0; st >>= 1) {
    if (tid < st) { r1[tid] += r1[tid + st]; r2[tid] += r2[tid + st]; }
    __syncthreads();
  }
  const float mean = r1[0] * (1.0f / D_MODEL);
  const float var  = r2[0] * (1.0f / D_MODEL) - mean * mean;
  const float inv  = rsqrtf(var + LN_EPS);
  _Float16* orow = out + (size_t)row * D_MODEL;
  #pragma unroll
  for (int i = tid; i < D_MODEL; i += 256)
    orow[i] = (_Float16)((xr[i] - mean) * inv * gam[i] + bet[i]);
}

// ---------------------------------------------------------------------------
// WMMA GEMM: C[M,N] = A[M,K] x W[N,K]^T + bias
//   MODE 0: f16 out          MODE 1: exact GELU -> f16 out
//   MODE 2: fp32 out = acc + bias + residual
// 128x128 block tile, K-step 32, double-buffered async LDS staging,
// 8 waves each owning 4x2 16x16 WMMA micro-tiles.
// ---------------------------------------------------------------------------
template <int MODE>
__global__ __launch_bounds__(256)
void gemm_wmma(const _Float16* __restrict__ A, const _Float16* __restrict__ W,
               const float* __restrict__ bias, const float* __restrict__ res,
               _Float16* __restrict__ Of16, float* __restrict__ Of32,
               int M, int N, int K) {
  __shared__ __align__(16) _Float16 As[2][128][40];  // 32 + pad (80B stride)
  __shared__ __align__(16) _Float16 Bs[2][128][40];

  const int tid  = threadIdx.x;
  const int wave = tid >> 5, lane = tid & 31;
  const int g = lane >> 4, ln = lane & 15;
  const int wm0 = (wave >> 2) * 64;   // wave row base inside tile (0/64)
  const int wn0 = (wave & 3) * 32;    // wave col base inside tile (0..96)
  const int m0 = blockIdx.y * 128, n0 = blockIdx.x * 128;

  const int lr = tid >> 1;            // loader row (0..127)
  const int lc = (tid & 1) * 16;      // loader half-row (0/16 halves)

  v8f acc[4][2] = {};

  auto stage = [&](int k0, int buf) {
    const _Float16* ga = A + (size_t)(m0 + lr) * K + k0 + lc;
    const _Float16* gw = W + (size_t)(n0 + lr) * K + k0 + lc;
    async_cp16(ga,     &As[buf][lr][lc]);
    async_cp16(ga + 8, &As[buf][lr][lc + 8]);
    async_cp16(gw,     &Bs[buf][lr][lc]);
    async_cp16(gw + 8, &Bs[buf][lr][lc + 8]);
  };

  const int nk = K >> 5;
  stage(0, 0);
  for (int i = 0; i < nk; ++i) {
    wait_async_all();          // my tile-i stores to LDS landed
    __syncthreads();           // everyone's landed; tile i-1 consumers done
    if (i + 1 < nk) stage((i + 1) << 5, (i + 1) & 1);  // prefetch tile i+1
    const int buf = i & 1;

    v16h af[4], bf[2];
    #pragma unroll
    for (int wm = 0; wm < 4; ++wm) {           // A frag: per-lane K map
      FragU f;
      f.u[0] = *(const uivec4*)&As[buf][wm0 + wm * 16 + ln][8 * g];
      f.u[1] = *(const uivec4*)&As[buf][wm0 + wm * 16 + ln][16 + 8 * g];
      af[wm] = f.h;
    }
    #pragma unroll
    for (int wn = 0; wn < 2; ++wn) {           // B frag: per-lane K map (same as A)
      FragU f;
      f.u[0] = *(const uivec4*)&Bs[buf][wn0 + wn * 16 + ln][8 * g];
      f.u[1] = *(const uivec4*)&Bs[buf][wn0 + wn * 16 + ln][16 + 8 * g];
      bf[wn] = f.h;
    }
    #pragma unroll
    for (int wm = 0; wm < 4; ++wm)
      #pragma unroll
      for (int wn = 0; wn < 2; ++wn)
        acc[wm][wn] = __builtin_amdgcn_wmma_f32_16x16x32_f16(
            false, af[wm], false, bf[wn], (short)0, acc[wm][wn], false, false);
  }

  #pragma unroll
  for (int wm = 0; wm < 4; ++wm)
    #pragma unroll
    for (int wn = 0; wn < 2; ++wn) {
      const int col = n0 + wn0 + wn * 16 + ln;
      const float bv = bias[col];
      #pragma unroll
      for (int rr = 0; rr < 8; ++rr) {
        const int row = m0 + wm0 + wm * 16 + rr + 8 * g;  // C/D: row = rr+8g
        const size_t idx = (size_t)row * N + col;
        float v = acc[wm][wn][rr] + bv;
        if constexpr (MODE == 0) {
          Of16[idx] = (_Float16)v;
        } else if constexpr (MODE == 1) {
          Of16[idx] = (_Float16)(0.5f * v * (1.0f + erff(v * 0.70710678118f)));
        } else {
          Of32[idx] = v + res[idx];
        }
      }
    }
}

// ---------------------------------------------------------------------------
// Flash attention: grid (S/128, H, B), 256 threads; each wave = 16 q rows.
// Q/K/V/O layout: [B*S, D_MODEL] f16, head h at cols h*64..h*64+63.
// K tiles staged async (double-buffered); V transposed manually into LDS.
// ---------------------------------------------------------------------------
__global__ __launch_bounds__(256)
void attn_flash(const _Float16* __restrict__ Q, const _Float16* __restrict__ Km,
                const _Float16* __restrict__ V, _Float16* __restrict__ O) {
  __shared__ __align__(16) _Float16 Ks[2][32][72];    // [key][dim], padded
  __shared__ __align__(16) _Float16 Vt[2][64][40];    // [dim][key], padded
  __shared__ __align__(16) _Float16 Ps[8][16][40];    // per-wave P relay

  const int tid = threadIdx.x, wave = tid >> 5, lane = tid & 31;
  const int g = lane >> 4, ln = lane & 15;
  const int q0 = blockIdx.x * 128;
  const int h = blockIdx.y, bz = blockIdx.z;
  const size_t base = (size_t)bz * SEQ * D_MODEL + (size_t)h * HD;
  const int qrow0 = q0 + wave * 16;

  // Resident Q fragments (two K-chunks of 32 over head_dim=64)
  v16h qf[2];
  #pragma unroll
  for (int t = 0; t < 2; ++t) {
    FragU f;
    const _Float16* qp = Q + base + (size_t)(qrow0 + ln) * D_MODEL + t * 32;
    f.u[0] = *(const uivec4*)(qp + 8 * g);
    f.u[1] = *(const uivec4*)(qp + 16 + 8 * g);
    qf[t] = f.h;
  }

  float mrow[8], lrow[8];
  v8f ctx[4] = {};
  #pragma unroll
  for (int r = 0; r < 8; ++r) { mrow[r] = -1e30f; lrow[r] = 0.f; }

  const int kr = tid >> 3, kc = (tid & 7) * 8;
  auto stageKV = [&](int j0, int buf) {
    async_cp16(Km + base + (size_t)(j0 + kr) * D_MODEL + kc, &Ks[buf][kr][kc]);
    union { uivec4 u; _Float16 hh[8]; } vv;
    vv.u = *(const uivec4*)(V + base + (size_t)(j0 + kr) * D_MODEL + kc);
    #pragma unroll
    for (int e = 0; e < 8; ++e) Vt[buf][kc + e][kr] = vv.hh[e];
  };

  const int ntiles = (q0 + 128) >> 5;   // causal upper bound, tiles of 32 keys
  stageKV(0, 0);
  for (int i = 0; i < ntiles; ++i) {
    wait_async_all();
    __syncthreads();
    if (i + 1 < ntiles) stageKV((i + 1) << 5, (i + 1) & 1);
    const int j0 = i << 5, buf = i & 1;

    // S = Q K^T : two 16-key sub-tiles, two 32-dim chunks each
    v8f sc[2] = {};
    #pragma unroll
    for (int sub = 0; sub < 2; ++sub)
      #pragma unroll
      for (int t = 0; t < 2; ++t) {
        FragU f;
        f.u[0] = *(const uivec4*)&Ks[buf][sub * 16 + ln][t * 32 + 8 * g];
        f.u[1] = *(const uivec4*)&Ks[buf][sub * 16 + ln][t * 32 + 16 + 8 * g];
        sc[sub] = __builtin_amdgcn_wmma_f32_16x16x32_f16(
            false, qf[t], false, f.h, (short)0, sc[sub], false, false);
      }

    const float scale = 0.125f;  // 1/sqrt(64)
    const int key0 = j0 + ln, key1 = j0 + 16 + ln;
    #pragma unroll
    for (int rr = 0; rr < 8; ++rr) {
      const int qr = qrow0 + rr + 8 * g;
      float v0 = (key0 <= qr) ? sc[0][rr] * scale : -1e9f;
      float v1 = (key1 <= qr) ? sc[1][rr] * scale : -1e9f;
      float rm = fmaxf(v0, v1);                 // row-max over 16 lanes
      #pragma unroll
      for (int m = 8; m >= 1; m >>= 1) rm = fmaxf(rm, __shfl_xor(rm, m, 32));
      const float mnew  = fmaxf(mrow[rr], rm);
      const float alpha = __expf(mrow[rr] - mnew);
      const float p0 = __expf(v0 - mnew), p1 = __expf(v1 - mnew);
      float rs = p0 + p1;                       // row-sum over 16 lanes
      #pragma unroll
      for (int m = 8; m >= 1; m >>= 1) rs += __shfl_xor(rs, m, 32);
      lrow[rr] = lrow[rr] * alpha + rs;
      mrow[rr] = mnew;
      #pragma unroll
      for (int db = 0; db < 4; ++db) ctx[db][rr] *= alpha;
      Ps[wave][rr + 8 * g][ln]      = (_Float16)p0;
      Ps[wave][rr + 8 * g][16 + ln] = (_Float16)p1;
    }

    // ctx += P V : A frag from per-wave LDS P buffer, B frags from Vt
    FragU pf;
    pf.u[0] = *(const uivec4*)&Ps[wave][ln][8 * g];
    pf.u[1] = *(const uivec4*)&Ps[wave][ln][16 + 8 * g];
    #pragma unroll
    for (int db = 0; db < 4; ++db) {
      FragU f;
      f.u[0] = *(const uivec4*)&Vt[buf][db * 16 + ln][8 * g];
      f.u[1] = *(const uivec4*)&Vt[buf][db * 16 + ln][16 + 8 * g];
      ctx[db] = __builtin_amdgcn_wmma_f32_16x16x32_f16(
          false, pf.h, false, f.h, (short)0, ctx[db], false, false);
    }
  }

  #pragma unroll
  for (int db = 0; db < 4; ++db)
    #pragma unroll
    for (int rr = 0; rr < 8; ++rr) {
      const int qr = qrow0 + rr + 8 * g;
      O[base + (size_t)qr * D_MODEL + db * 16 + ln] =
          (_Float16)(ctx[db][rr] / lrow[rr]);
    }
}

// ---------------------------------------------------------------------------
// Host orchestration
// ---------------------------------------------------------------------------
extern "C" void kernel_launch(void* const* d_in, const int* in_sizes, int n_in,
                              void* d_out, int out_size, void* d_ws,
                              size_t ws_size, hipStream_t stream) {
  (void)in_sizes; (void)n_in; (void)out_size; (void)ws_size;
  const float* x   = (const float*)d_in[0];
  // d_in[1] = causal mask (hardcoded in attention kernel)
  const float* Wq  = (const float*)d_in[2];
  const float* bq  = (const float*)d_in[3];
  const float* Wk  = (const float*)d_in[4];
  const float* bk  = (const float*)d_in[5];
  const float* Wv  = (const float*)d_in[6];
  const float* bv  = (const float*)d_in[7];
  const float* Wo  = (const float*)d_in[8];
  const float* bo  = (const float*)d_in[9];
  const float* g1  = (const float*)d_in[10];
  const float* be1 = (const float*)d_in[11];
  const float* g2  = (const float*)d_in[12];
  const float* be2 = (const float*)d_in[13];
  const float* W1  = (const float*)d_in[14];
  const float* b1  = (const float*)d_in[15];
  const float* W2  = (const float*)d_in[16];
  const float* b2  = (const float*)d_in[17];

  const int DD = D_MODEL * D_MODEL;   // 1M
  const int DF = D_FF * D_MODEL;      // 4M

  char* p = (char*)d_ws;
  auto alloc = [&](size_t bytes) {
    char* r = p; p += (bytes + 255) & ~(size_t)255; return r;
  };
  _Float16* Wq16  = (_Float16*)alloc((size_t)DD * 2);
  _Float16* Wk16  = (_Float16*)alloc((size_t)DD * 2);
  _Float16* Wv16  = (_Float16*)alloc((size_t)DD * 2);
  _Float16* Wo16  = (_Float16*)alloc((size_t)DD * 2);
  _Float16* W116  = (_Float16*)alloc((size_t)DF * 2);
  _Float16* W216  = (_Float16*)alloc((size_t)DF * 2);
  _Float16* h16   = (_Float16*)alloc((size_t)ROWS * D_MODEL * 2);
  _Float16* q16   = (_Float16*)alloc((size_t)ROWS * D_MODEL * 2);
  _Float16* k16   = (_Float16*)alloc((size_t)ROWS * D_MODEL * 2);
  _Float16* v16   = (_Float16*)alloc((size_t)ROWS * D_MODEL * 2);
  _Float16* ctx16 = (_Float16*)alloc((size_t)ROWS * D_MODEL * 2);
  float*    x1    = (float*)   alloc((size_t)ROWS * D_MODEL * 4);
  _Float16* ff16  = (_Float16*)alloc((size_t)ROWS * D_FF * 2);

  // Weight conversion fp32 -> f16
  cvt_f32_f16<<<512,  256, 0, stream>>>(Wq, Wq16, DD);
  cvt_f32_f16<<<512,  256, 0, stream>>>(Wk, Wk16, DD);
  cvt_f32_f16<<<512,  256, 0, stream>>>(Wv, Wv16, DD);
  cvt_f32_f16<<<512,  256, 0, stream>>>(Wo, Wo16, DD);
  cvt_f32_f16<<<2048, 256, 0, stream>>>(W1, W116, DF);
  cvt_f32_f16<<<2048, 256, 0, stream>>>(W2, W216, DF);

  const dim3 gD(D_MODEL / 128, ROWS / 128);  // (8, 32)
  const dim3 gF(D_FF    / 128, ROWS / 128);  // (32, 32)

  // h = LN1(x)
  layernorm_f16<<<ROWS, 256, 0, stream>>>(x, g1, be1, h16);
  // q,k,v = h @ W^T + b
  gemm_wmma<0><<<gD, 256, 0, stream>>>(h16, Wq16, bq, nullptr, q16, nullptr,
                                       ROWS, D_MODEL, D_MODEL);
  gemm_wmma<0><<<gD, 256, 0, stream>>>(h16, Wk16, bk, nullptr, k16, nullptr,
                                       ROWS, D_MODEL, D_MODEL);
  gemm_wmma<0><<<gD, 256, 0, stream>>>(h16, Wv16, bv, nullptr, v16, nullptr,
                                       ROWS, D_MODEL, D_MODEL);
  // ctx = causal softmax(q k^T / 8) v
  attn_flash<<<dim3(SEQ / 128, HEADS, BATCH), 256, 0, stream>>>(q16, k16, v16,
                                                                ctx16);
  // x1 = x + ctx @ Wo^T + bo
  gemm_wmma<2><<<gD, 256, 0, stream>>>(ctx16, Wo16, bo, x, nullptr, x1,
                                       ROWS, D_MODEL, D_MODEL);
  // h = LN2(x1)
  layernorm_f16<<<ROWS, 256, 0, stream>>>(x1, g2, be2, h16);
  // ff = gelu(h @ W1^T + b1)
  gemm_wmma<1><<<gF, 256, 0, stream>>>(h16, W116, b1, nullptr, ff16, nullptr,
                                       ROWS, D_FF, D_MODEL);
  // out = x1 + ff @ W2^T + b2
  gemm_wmma<2><<<gD, 256, 0, stream>>>(ff16, W216, b2, x1, nullptr,
                                       (float*)d_out, ROWS, D_MODEL, D_FF);
}
